// DilatedReparamBlock_21139829031531
// MI455X (gfx1250) — compile-verified, matrix operand order checked by reading, not run on hardware
//
#include <hip/hip_runtime.h>

// ---------------------------------------------------------------------------
// DilatedReparamBlock on MI455X (gfx1250):
//   (1) fold 7 depthwise branches + BN into ONE fused 13x13 dw kernel + bias
//   (2) run the fused conv as bf16 WMMA (v_wmma_f32_16x16x32_bf16) via a
//       Toeplitz-along-x mapping, f32 accumulate, LDS-staged input tiles.
// Memory-bound target: ~205 MB @ 23.3 TB/s ~= 9 us; this round trims the
// per-thread issue-slot overhead (shared B-fragment table, vectorized fill).
// ---------------------------------------------------------------------------

typedef __attribute__((ext_vector_type(16))) __bf16 v16bf;
typedef __attribute__((ext_vector_type(8)))  __bf16 v8bf;
typedef __attribute__((ext_vector_type(8)))  float  v8f;
typedef __attribute__((ext_vector_type(4)))  float  v4f;

#define CCH   256
#define IH    56
#define IW    56
#define PAD   6
#define LROWS 68          // padded rows actually touched (0..67)
#define LSTR  88          // halves per LDS row (176 B: 16B-aligned, bank-clean)
#define WSTRIDE 170       // 169 fused taps + 1 bias per channel

static __device__ inline unsigned short f2bfu(float f) {
  // round-to-nearest-even f32 -> bf16 raw bits
  unsigned u = __float_as_uint(f);
  unsigned r = (u + 0x7FFFu + ((u >> 16) & 1u)) >> 16;
  return (unsigned short)r;
}
static __device__ inline __bf16 f2bf(float f) {
  unsigned short s = f2bfu(f);
  return __builtin_bit_cast(__bf16, s);
}

// ---------------------------------------------------------------------------
// Kernel 1: build fused 13x13 weights + bias per channel.
// ---------------------------------------------------------------------------
__global__ void fuse_weights_kernel(const float* __restrict__ lk_w,
                                    const float* __restrict__ w0,
                                    const float* __restrict__ w1,
                                    const float* __restrict__ w2,
                                    const float* __restrict__ w3,
                                    const float* __restrict__ w4,
                                    const float* __restrict__ w5,
                                    const float* __restrict__ bn_g,
                                    const float* __restrict__ bn_b,
                                    const float* __restrict__ bn_m,
                                    const float* __restrict__ bn_v,
                                    float* __restrict__ fused) {
  const int c = blockIdx.x;     // channel
  const int t = threadIdx.x;    // 0..255
  __shared__ float scale[7], shift[7];
  if (t < 7) {
    float g  = bn_g[t * CCH + c];
    float be = bn_b[t * CCH + c];
    float mu = bn_m[t * CCH + c];
    float va = bn_v[t * CCH + c];
    float s = g * rsqrtf(va + 1e-5f);
    scale[t] = s;
    shift[t] = be - mu * s;
  }
  __syncthreads();

  if (t < 169) {
    const int p = t / 13, q = t % 13;
    float acc = scale[0] * lk_w[c * 169 + t];
    const float* wbr[6] = {w0, w1, w2, w3, w4, w5};
    const int K_[6] = {5, 7, 7, 3, 3, 3};
    const int R_[6] = {1, 1, 2, 3, 4, 5};
#pragma unroll
    for (int i = 0; i < 6; ++i) {
      const int k = K_[i], r = R_[i];
      const int e = r * (k - 1) + 1;     // effective extent (odd, <= 13)
      const int o = (13 - e) / 2;        // centering offset
      const int pp = p - o, qq = q - o;
      if (pp >= 0 && qq >= 0 && (pp % r) == 0 && (qq % r) == 0) {
        const int u = pp / r, v = qq / r;
        if (u < k && v < k)
          acc += scale[i + 1] * wbr[i][(c * k + u) * k + v];
      }
    }
    fused[c * WSTRIDE + t] = acc;
  } else if (t == 169) {
    float b = 0.f;
#pragma unroll
    for (int i = 0; i < 7; ++i) b += shift[i];
    fused[c * WSTRIDE + 169] = b;
  }
}

// ---------------------------------------------------------------------------
// Kernel 2: fused 13x13 depthwise conv via bf16 WMMA.
// One workgroup per (batch, channel) image. 8 waves, 16 tiles (4x4 grid with
// overlapping edge tiles at 40 -> no bounds checks, deterministic rewrites).
//
// Per tile, per ky:  D[y][x] += A[y][j] * B[j][x]
//   A[y][j] = in_pad[y0+y+ky][x0+j]         (16x32 bf16, LDS b128 loads)
//   B[j][x] = wfused[ky][j-x] (Toeplitz)    (32x16 bf16, table built once
//                                            per workgroup in LDS)
// ---------------------------------------------------------------------------
__global__ __launch_bounds__(256, 1)
void dwconv13_wmma_kernel(const float* __restrict__ x,
                          const float* __restrict__ fused,
                          float* __restrict__ out) {
  __shared__ __bf16 simg[LROWS * LSTR];     // 11968 B padded bf16 image
  __shared__ __bf16 sB[13 * 32 * 16];       // 13312 B Toeplitz B-frag table
  __shared__ float  swt[WSTRIDE];           // fused taps + bias

  const int bc   = blockIdx.x;              // b*256 + c
  const int c    = bc & (CCH - 1);
  const int tid  = threadIdx.x;
  const int lane = tid & 31;
  const int wv   = tid >> 5;                // wave id 0..7
  const int hi   = lane >> 4;               // half-wave selector
  const int n15  = lane & 15;

  if (tid < WSTRIDE) swt[tid] = fused[c * WSTRIDE + tid];

  // zero the padded LDS image (as dwords)
  {
    unsigned int* z = (unsigned int*)simg;
    for (int i = tid; i < (LROWS * LSTR) / 2; i += 256) z[i] = 0u;
  }
  __syncthreads();   // swt + zeroed image visible

  // ---- fill interior: vectorized f32x4 global -> packed bf16 LDS
  // 56 cols = 14 float4 per row; fully coalesced b128 loads.
  const float* src = x + (size_t)bc * (IH * IW);
  for (int i4 = tid; i4 < IH * (IW / 4); i4 += 256) {
    const int h = i4 / (IW / 4);
    const int q = i4 - h * (IW / 4);
    const int w = q * 4;
    v4f v = *(const v4f*)(src + h * IW + w);          // 16B aligned
    unsigned lo = ((unsigned)f2bfu(v.y) << 16) | f2bfu(v.x);
    unsigned hi32 = ((unsigned)f2bfu(v.w) << 16) | f2bfu(v.z);
    unsigned* p = (unsigned*)&simg[(h + PAD) * LSTR + (PAD + w)]; // 4B aligned
    p[0] = lo;
    p[1] = hi32;
  }

  // ---- build Toeplitz B table cooperatively (once per workgroup).
  // Entry (ky, l, e): register element e of lane l for tap-row ky.
  // 16-bit operand layout: element e -> VGPR e/2 half e%2;
  // K(e,l) = ((e<8) ? e : e+8) + 8*(l>>4) ; N = l&15 ; B[K][N] = w[ky][K-N]
  for (int idx = tid; idx < 13 * 32 * 16; idx += 256) {
    const int e  = idx & 15;
    const int l  = (idx >> 4) & 31;
    const int ky = idx >> 9;
    const int K  = ((e < 8) ? e : e + 8) + 8 * (l >> 4);
    const int kx = K - (l & 15);
    float wval = 0.f;
    if (kx >= 0 && kx <= 12) wval = swt[ky * 13 + kx];
    sB[idx] = f2bf(wval);
  }
  __syncthreads();

  // ---- materialize this wave's 13 B fragments (2x ds_load_b128 each)
  v16bf bfr[13];
#pragma unroll
  for (int ky = 0; ky < 13; ++ky)
    bfr[ky] = *(const v16bf*)&sB[(ky * 32 + lane) * 16];   // 32B aligned

  const float bias = swt[169];
  float* dst = out + (size_t)bc * (IH * IW);

  // ---- 16 tiles (4x4), 2 per wave; edge tiles overlap (40..55)
  for (int t = wv; t < 16; t += 8) {
    const int ty = t >> 2, tx = t & 3;
    const int y0 = (ty == 3) ? 40 : ty * 16;
    const int x0 = (tx == 3) ? 40 : tx * 16;

    v8f acc = {};
#pragma unroll
    for (int ky = 0; ky < 13; ++ky) {
      // A fragment: lane row = y0 + (lane&15) + ky; two 8-half chunks
      // elements 0..7  -> cols x0 + 8*hi + (0..7)
      // elements 8..15 -> cols x0 + 8*hi + 16 + (0..7)
      const int rA   = y0 + n15 + ky;
      const int cb   = x0 + hi * 8;
      const v8bf* ap = (const v8bf*)&simg[rA * LSTR + cb];   // 16B aligned
      v8bf a0 = ap[0];
      v8bf a1 = ap[2];                                       // +16 halves
      v16bf av;
#pragma unroll
      for (int e = 0; e < 8; ++e) { av[e] = a0[e]; av[e + 8] = a1[e]; }

      acc = __builtin_amdgcn_wmma_f32_16x16x32_bf16(
          /*neg_a=*/false, av, /*neg_b=*/false, bfr[ky],
          /*c_mod=*/(short)0, acc, /*reuse_a=*/false, /*reuse_b=*/false);
    }

    // D layout: lane<16 -> M=r, N=lane ; lane>=16 -> M=8+r, N=lane-16
#pragma unroll
    for (int r = 0; r < 8; ++r) {
      const int row = y0 + hi * 8 + r;
      dst[row * IW + x0 + n15] = acc[r] + bias;
    }
  }
}

// ---------------------------------------------------------------------------
extern "C" void kernel_launch(void* const* d_in, const int* in_sizes, int n_in,
                              void* d_out, int out_size, void* d_ws, size_t ws_size,
                              hipStream_t stream) {
  const float* x    = (const float*)d_in[0];
  const float* lk_w = (const float*)d_in[1];
  const float* w0   = (const float*)d_in[2];
  const float* w1   = (const float*)d_in[3];
  const float* w2   = (const float*)d_in[4];
  const float* w3   = (const float*)d_in[5];
  const float* w4   = (const float*)d_in[6];
  const float* w5   = (const float*)d_in[7];
  const float* bn_g = (const float*)d_in[8];
  const float* bn_b = (const float*)d_in[9];
  const float* bn_m = (const float*)d_in[10];
  const float* bn_v = (const float*)d_in[11];
  float* fused = (float*)d_ws;                 // 256*170 floats of scratch
  float* outp  = (float*)d_out;

  fuse_weights_kernel<<<CCH, 256, 0, stream>>>(lk_w, w0, w1, w2, w3, w4, w5,
                                               bn_g, bn_b, bn_m, bn_v, fused);
  dwconv13_wmma_kernel<<<32 * CCH, 256, 0, stream>>>(x, fused, outp);
}
